// DisorderedCausalSelfAttention_25683904430760
// MI455X (gfx1250) — matrix operand order repaired and checked by reading, not compiled, not run on hardware
//
#include <hip/hip_runtime.h>

typedef _Float16 h16;
typedef __attribute__((ext_vector_type(16))) _Float16 v16h;
typedef __attribute__((ext_vector_type(8)))  _Float16 v8h;
typedef __attribute__((ext_vector_type(8)))  float    v8f;
typedef __attribute__((ext_vector_type(2)))  int      v2i;

static constexpr int Bb = 4, Tt = 2048, Cc = 1024, NHh = 16, Dd = 64;
static constexpr int Mrows = Bb * Tt;   // 8192

#define WMMA_F16(a, b, c) \
  __builtin_amdgcn_wmma_f32_16x16x32_f16(false, (a), false, (b), (short)0, (c), false, false)

// gfx1250 async global->LDS copy path (ASYNCcnt), guarded so either toolchain
// falls back to a synchronous staging copy if the builtins are absent.
#if defined(__has_builtin)
#if __has_builtin(__builtin_amdgcn_global_load_async_to_lds_b64) && \
    __has_builtin(__builtin_amdgcn_s_wait_asynccnt)
#define HAVE_ASYNC_LDS 1
#endif
#endif
#ifndef HAVE_ASYNC_LDS
#define HAVE_ASYNC_LDS 0
#endif

#if HAVE_ASYNC_LDS
typedef __attribute__((address_space(1))) v2i gv2i;   // global 64-bit payload
typedef __attribute__((address_space(3))) v2i lv2i;   // LDS 64-bit payload
#endif

// ---------------------------------------------------------------- f32 -> f16
__global__ void cvt_f32_f16(const float* __restrict__ src, h16* __restrict__ dst, int n) {
  int i = blockIdx.x * blockDim.x + threadIdx.x;
  int stride = gridDim.x * blockDim.x;
  for (; i < n; i += stride) dst[i] = (h16)src[i];
}

// B-matrix fragment loader: 16 contiguous halfs from ptr -> v16h
__device__ __forceinline__ v16h load_bfrag(const h16* ptr) {
  v8h x0 = *reinterpret_cast<const v8h*>(ptr);
  v8h x1 = *reinterpret_cast<const v8h*>(ptr + 8);
  v16h r;
#pragma unroll
  for (int i = 0; i < 8; ++i) { r[i] = x0[i]; r[i + 8] = x1[i]; }
  return r;
}

// ------------------------------------------------------- tiled WMMA GEMM
// MODE 0: qkv = xh @ W_attn ; epilogue adds b_attn (+bQ,*1/8 | +bK), scatters
//         Q,K head-major [B,NH,T,D]; V transposed head-major [B,NH,D,T].
// MODE 1: out = Yh @ W_proj + b_proj (f32).
template <int MODE>
__launch_bounds__(128)
__global__ void gemm16x64(const h16* __restrict__ A, const h16* __restrict__ Bm,
                          const float* __restrict__ bias,
                          const float* __restrict__ bQ, const float* __restrict__ bK,
                          h16* __restrict__ Qh, h16* __restrict__ Kh, h16* __restrict__ Vt,
                          float* __restrict__ out) {
  constexpr int N = (MODE == 0) ? 3 * Cc : Cc;
  constexpr int K = Cc;
  constexpr int BSTR = 48;  // padded LDS row stride (halfs) for transposed B tile

  __shared__ h16 Atile[16 * 32];
  __shared__ h16 BtileT[64 * BSTR];

  const int tid    = threadIdx.x;
  const int lane   = tid & 31;
  const int wave   = tid >> 5;
  const int laneHi = (lane >> 4) & 1;
  const int lm     = lane & 15;

  const int rowBase = blockIdx.x * 16;
  const int colBase = blockIdx.y * 64;

  v8f acc = {};

  for (int kt = 0; kt < K / 32; ++kt) {
    { // stage A tile (16x32): 8 bytes / thread, async DMA into LDS when available
      int idx = tid * 4;
      int r = idx >> 5, c = idx & 31;
      const h16* src = A + (size_t)(rowBase + r) * K + kt * 32 + c;
#if HAVE_ASYNC_LDS
      __builtin_amdgcn_global_load_async_to_lds_b64(
          (gv2i*)(void*)src, (lv2i*)(void*)&Atile[idx], 0, 0);
#else
      *reinterpret_cast<unsigned long long*>(&Atile[idx]) =
          *reinterpret_cast<const unsigned long long*>(src);
#endif
    }
    { // stage B tile (32x64) transposed into LDS: 16 halfs / thread
      int r = tid >> 2, c = (tid & 3) * 16;
      const h16* src = Bm + (size_t)(kt * 32 + r) * N + colBase + c;
      v8h b0 = *reinterpret_cast<const v8h*>(src);
      v8h b1 = *reinterpret_cast<const v8h*>(src + 8);
#pragma unroll
      for (int i = 0; i < 8; ++i) {
        BtileT[(c + i) * BSTR + r]     = b0[i];
        BtileT[(c + 8 + i) * BSTR + r] = b1[i];
      }
    }
#if HAVE_ASYNC_LDS
    __builtin_amdgcn_s_wait_asynccnt(0);  // drain this wave's async LDS writes
#endif
    __syncthreads();

    v16h a, b;
    { // A fragment (16x32 layout)
      const h16* pa = &Atile[lm * 32 + laneHi * 8];
      v8h a0 = *reinterpret_cast<const v8h*>(pa);
      v8h a1 = *reinterpret_cast<const v8h*>(pa + 16);
#pragma unroll
      for (int i = 0; i < 8; ++i) { a[i] = a0[i]; a[i + 8] = a1[i]; }
    }
    { // B fragment: column = wave*16 + lm, contiguous in transposed tile
      const h16* pb = &BtileT[(wave * 16 + lm) * BSTR + laneHi * 8];
      v8h b0 = *reinterpret_cast<const v8h*>(pb);
      v8h b1 = *reinterpret_cast<const v8h*>(pb + 16);
#pragma unroll
      for (int i = 0; i < 8; ++i) { b[i] = b0[i]; b[i + 8] = b1[i]; }
    }
    acc = WMMA_F16(a, b, acc);
    __syncthreads();
  }

  const int col = colBase + wave * 16 + lm;
#pragma unroll
  for (int v = 0; v < 8; ++v) {
    const int row = rowBase + v + 8 * laneHi;
    float val = acc[v];
    if (MODE == 0) {
      int which = col >> 10;          // 0=q 1=k 2=v
      int c  = col & (Cc - 1);
      int hh = c >> 6, d = c & 63;
      val += bias[col];
      int bb = row >> 11, t = row & (Tt - 1);
      if (which == 0) {               // fold 1/sqrt(64) into Q
        val = (val + bQ[hh * 64 + d]) * 0.125f;
        Qh[(size_t)((bb * NHh + hh) * Tt + t) * Dd + d] = (h16)val;
      } else if (which == 1) {
        val += bK[hh * 64 + d];
        Kh[(size_t)((bb * NHh + hh) * Tt + t) * Dd + d] = (h16)val;
      } else {                        // V stored transposed: [B,NH,D,T]
        Vt[(size_t)((bb * NHh + hh) * Dd + d) * Tt + t] = (h16)val;
      }
    } else {
      out[(size_t)row * Cc + col] = val + bias[col];
    }
  }
}

// -------------------------------------------------- flash causal attention
// 1 wave per (b, h, 16-row q tile); 32 keys per iteration.
__launch_bounds__(32)
__global__ void attn_fwd(const h16* __restrict__ Qh, const h16* __restrict__ Kh,
                         const h16* __restrict__ Vt, h16* __restrict__ Yh) {
  __shared__ h16 Plds[16 * 32];

  const int lane   = threadIdx.x & 31;
  const int laneHi = (lane >> 4) & 1;
  const int lm     = lane & 15;
  const int qt = blockIdx.x, h = blockIdx.y, b = blockIdx.z;
  const int qbase = qt * 16;

  const size_t headOff = (size_t)(b * NHh + h) * Tt * Dd;
  const h16* Q  = Qh + headOff;
  const h16* Kp = Kh + headOff;
  const h16* Vp = Vt + headOff;   // [D][T] per head

  // Q fragments for D=64 (two K=32 steps), kept in registers.
  v16h aq0, aq1;
  {
    const h16* qrow = Q + (size_t)(qbase + lm) * Dd;
    v8h t0 = *reinterpret_cast<const v8h*>(qrow + laneHi * 16);
    v8h t1 = *reinterpret_cast<const v8h*>(qrow + 8 + laneHi * 16);
    v8h t2 = *reinterpret_cast<const v8h*>(qrow + 32 + laneHi * 16);
    v8h t3 = *reinterpret_cast<const v8h*>(qrow + 40 + laneHi * 16);
#pragma unroll
    for (int i = 0; i < 8; ++i) {
      aq0[i] = t0[i]; aq0[i + 8] = t1[i];
      aq1[i] = t2[i]; aq1[i + 8] = t3[i];
    }
  }

  v8f o[4] = {v8f{}, v8f{}, v8f{}, v8f{}};
  float Mr[8], Lr[8];
#pragma unroll
  for (int v = 0; v < 8; ++v) { Mr[v] = -1e30f; Lr[v] = 0.0f; }

  const int nblocks = qt / 2 + 1;
  for (int jb = 0; jb < nblocks; ++jb) {
    const int jbase = jb * 32;

    // S = Q K^T for 32 keys (two 16-col WMMA tiles, two K-steps each)
    const h16* k0row = Kp + (size_t)(jbase + lm) * Dd;       // key tile 0 column = lm
    const h16* k1row = Kp + (size_t)(jbase + 16 + lm) * Dd;  // key tile 1
    v8f s0 = {}, s1 = {};
    s0 = WMMA_F16(aq0, load_bfrag(k0row + laneHi * 16), s0);
    s0 = WMMA_F16(aq1, load_bfrag(k0row + 32 + laneHi * 16), s0);
    s1 = WMMA_F16(aq0, load_bfrag(k1row + laneHi * 16), s1);
    s1 = WMMA_F16(aq1, load_bfrag(k1row + 32 + laneHi * 16), s1);

    if (jb == nblocks - 1) {  // causal mask (only last block can cross diagonal)
#pragma unroll
      for (int v = 0; v < 8; ++v) {
        int qi = qbase + v + 8 * laneHi;
        if (jbase + lm > qi)      s0[v] = -1e30f;
        if (jbase + 16 + lm > qi) s1[v] = -1e30f;
      }
    }

    // online softmax: row m = v + 8*laneHi lives across 16 lanes of one half-wave
#pragma unroll
    for (int v = 0; v < 8; ++v) {
      float mv = fmaxf(s0[v], s1[v]);
      mv = fmaxf(mv, __shfl_xor(mv, 1, 32));
      mv = fmaxf(mv, __shfl_xor(mv, 2, 32));
      mv = fmaxf(mv, __shfl_xor(mv, 4, 32));
      mv = fmaxf(mv, __shfl_xor(mv, 8, 32));
      float nm = fmaxf(Mr[v], mv);
      float sc = __expf(Mr[v] - nm);
      Mr[v] = nm;
      float p0 = __expf(s0[v] - nm);
      float p1 = __expf(s1[v] - nm);
      s0[v] = p0; s1[v] = p1;
      float rs = p0 + p1;
      rs += __shfl_xor(rs, 1, 32);
      rs += __shfl_xor(rs, 2, 32);
      rs += __shfl_xor(rs, 4, 32);
      rs += __shfl_xor(rs, 8, 32);
      Lr[v] = Lr[v] * sc + rs;
      o[0][v] *= sc; o[1][v] *= sc; o[2][v] *= sc; o[3][v] *= sc;
    }

    // transpose P (C-layout -> A-layout) through LDS
    __syncthreads();
#pragma unroll
    for (int v = 0; v < 8; ++v) {
      Plds[(v + 8 * laneHi) * 32 + lm]      = (h16)s0[v];
      Plds[(v + 8 * laneHi) * 32 + 16 + lm] = (h16)s1[v];
    }
    __syncthreads();
    v16h ap;
    {
      const h16* pp = &Plds[lm * 32 + laneHi * 16];
      v8h p0v = *reinterpret_cast<const v8h*>(pp);
      v8h p1v = *reinterpret_cast<const v8h*>(pp + 8);
#pragma unroll
      for (int i = 0; i < 8; ++i) { ap[i] = p0v[i]; ap[i + 8] = p1v[i]; }
    }

    // O += P @ V  (K = 32 keys, N = 64 = 4 WMMA tiles)
    // V transposed per head: B-frag for output col d is contiguous in [d][t]
#pragma unroll
    for (int t = 0; t < 4; ++t) {
      const h16* vrow = Vp + (size_t)(t * 16 + lm) * Tt + jbase + laneHi * 16;
      o[t] = WMMA_F16(ap, load_bfrag(vrow), o[t]);
    }
  }

  // normalize and store as f16 in [B,T,C] for the projection GEMM
#pragma unroll
  for (int v = 0; v < 8; ++v) {
    float inv = 1.0f / Lr[v];
    int trow = qbase + v + 8 * laneHi;
    h16* yrow = Yh + (size_t)(b * Tt + trow) * Cc + h * Dd;
#pragma unroll
    for (int t = 0; t < 4; ++t) yrow[t * 16 + lm] = (h16)(o[t][v] * inv);
  }
}

// ------------------------------------------------------------------ launch
extern "C" void kernel_launch(void* const* d_in, const int* in_sizes, int n_in,
                              void* d_out, int out_size, void* d_ws, size_t ws_size,
                              hipStream_t stream) {
  const float* x      = (const float*)d_in[0];
  const float* W_attn = (const float*)d_in[1];
  const float* b_attn = (const float*)d_in[2];
  const float* bQ     = (const float*)d_in[3];
  const float* bK     = (const float*)d_in[4];
  const float* W_proj = (const float*)d_in[5];
  const float* b_proj = (const float*)d_in[6];
  float* out = (float*)d_out;

  char* ws = (char*)d_ws;
  h16* xh  = (h16*)(ws);              // 16 MB  (B*T*C)
  h16* Wah = (h16*)(ws + 16777216);   // 6 MB   (C*3C)
  h16* Wph = (h16*)(ws + 23068672);   // 2 MB   (C*C)
  h16* Qh  = (h16*)(ws + 25165824);   // 16 MB  [B,NH,T,D]
  h16* Kh  = (h16*)(ws + 41943040);   // 16 MB  [B,NH,T,D]
  h16* Vt  = (h16*)(ws + 58720256);   // 16 MB  [B,NH,D,T] (transposed)
  h16* Yh  = (h16*)(ws + 75497472);   // 16 MB  [B,T,C]

  cvt_f32_f16<<<4096, 256, 0, stream>>>(x, xh, Bb * Tt * Cc);
  cvt_f32_f16<<<4096, 256, 0, stream>>>(W_attn, Wah, Cc * 3 * Cc);
  cvt_f32_f16<<<2048, 256, 0, stream>>>(W_proj, Wph, Cc * Cc);

  gemm16x64<0><<<dim3(Mrows / 16, (3 * Cc) / 64), 128, 0, stream>>>(
      xh, Wah, b_attn, bQ, bK, Qh, Kh, Vt, nullptr);

  attn_fwd<<<dim3(Tt / 16, NHh, Bb), 32, 0, stream>>>(Qh, Kh, Vt, Yh);

  gemm16x64<1><<<dim3(Mrows / 16, Cc / 64), 128, 0, stream>>>(
      Yh, Wph, b_proj, nullptr, nullptr, nullptr, nullptr, nullptr, out);
}
